// Attention_76725295776417
// MI455X (gfx1250) — matrix operand order repaired and checked by reading, not compiled, not run on hardware
//
#include <hip/hip_runtime.h>
#include <hip/hip_bf16.h>
#include <math.h>

// ---------------------------------------------------------------------------
// Problem constants (match the reference)
// ---------------------------------------------------------------------------
#define BATCH 2
#define SEQ   2048
#define DMODEL 4096
#define QHEADS 32
#define KVHEADS 8
#define HEADDIM 128
#define KVDIM (KVHEADS * HEADDIM)   // 1024
#define GROUP (QHEADS / KVHEADS)    // 4

typedef __attribute__((ext_vector_type(16))) __bf16 v16bf;
typedef __attribute__((ext_vector_type(8)))  float  v8f;
typedef __attribute__((ext_vector_type(4)))  unsigned v4u;
typedef __attribute__((ext_vector_type(8)))  unsigned v8u;

static __device__ __forceinline__ __bf16 f2bf(float f) { return (__bf16)f; }

// ---------------------------------------------------------------------------
// CDNA5 async global->LDS copy (ASYNCcnt-tracked), 16B per lane.
// Generic pointers to LDS hold the LDS byte offset in their low 32 bits
// (aperture mapping), which is what the VDST operand of the async op wants.
// ---------------------------------------------------------------------------
static __device__ __forceinline__ void async_copy_b128(void* lds_dst,
                                                       const void* gsrc) {
  const unsigned lds = (unsigned)(uintptr_t)lds_dst;
  const unsigned long long ga = (unsigned long long)(uintptr_t)gsrc;
  asm volatile("global_load_async_to_lds_b128 %0, %1, off"
               :
               : "v"(lds), "v"(ga)
               : "memory");
}
static __device__ __forceinline__ void wait_asynccnt0() {
  asm volatile("s_wait_asynccnt 0x0" ::: "memory");
}

// ---------------------------------------------------------------------------
// CDNA5 Tensor Data Mover: 2D tile (bf16) global -> LDS.
// D# built in SGPRs per CDNA5 ISA:
//   group0: [1:0]=count(1) | lds_addr[63:32] | global_addr[120:64] | type=2
//   group1: data_size=1(2B), tensor_dim0/1, tile_dim0/1, tensor_dim0_stride
//   group2/3: zero (2D tile)
// Issue from ONE wave only (TDM ignores EXEC); completion = TENSORcnt.
// ---------------------------------------------------------------------------
static __device__ __forceinline__ void tdm_load_2d_bf16(
    void* lds_dst, const void* gsrc, unsigned tile_w, unsigned tile_h,
    unsigned row_stride_elems) {
  const unsigned lds = (unsigned)(uintptr_t)lds_dst;
  const unsigned long long ga = (unsigned long long)(uintptr_t)gsrc;
  v4u g0, g2 = {0u, 0u, 0u, 0u}, g3 = {0u, 0u, 0u, 0u};
  v8u g1;
  g0[0] = 1u;                                   // count=1 (valid descriptor)
  g0[1] = lds;                                  // lds_addr
  g0[2] = (unsigned)ga;                         // global_addr[31:0]
  g0[3] = (unsigned)((ga >> 32) & 0x01ffffffu)  // global_addr[56:32]
          | 0x80000000u;                        // type=2 ("image")
  g1[0] = 0x00010000u;                          // data_size=1 (2 bytes)
  g1[1] = (tile_w & 0xffffu) << 16;             // tensor_dim0[15:0]
  g1[2] = (tile_w >> 16) | ((tile_h & 0xffffu) << 16);  // dim0 hi, dim1 lo
  g1[3] = (tile_h >> 16) | ((tile_w & 0xffffu) << 16);  // dim1 hi, tile_dim0
  g1[4] = tile_h & 0xffffu;                     // tile_dim1 (tile_dim2=0)
  g1[5] = row_stride_elems;                     // tensor_dim0_stride[31:0]
  g1[6] = 0u;                                   // stride hi, dim1_stride lo
  g1[7] = 0u;
  asm volatile("tensor_load_to_lds %0, %1, %2, %3"
               :
               : "s"(g0), "s"(g1), "s"(g2), "s"(g3)
               : "memory");
}

// ---------------------------------------------------------------------------
// GEMM:  Out[m,n] = sum_k A[m,k] * W[n,k] + bias[n]
//   A: (M,K) row-major, f32 or bf16 (template)
//   W: (N,K) row-major f32 (weights as given: y = x @ W^T)
//   Out: bf16 or f32 (template)
// 256 threads (8 waves). Tile M=128, N=128, K-step 32, LDS double-buffered:
// tile ks+1 is staged (TDM for bf16 A / convert path for f32) while tile ks
// feeds 8 WMMAs per wave (2 A-frags x 4 B-frags; B-frag reused by 2 WMMAs).
// M, N, K are multiples of 128/128/32 for every call here -> no bounds checks.
// ---------------------------------------------------------------------------
template <bool A_IS_BF16, bool OUT_BF16>
__global__ __launch_bounds__(256) void gemm_wt_kernel(
    const void* __restrict__ Aptr, const float* __restrict__ W,
    const float* __restrict__ bias, void* __restrict__ Out,
    int M, int N, int K) {
  __shared__ __bf16 As[2][128][32];
  __shared__ __bf16 Ws[2][128][32];

  const int m0 = blockIdx.x * 128;
  const int n0 = blockIdx.y * 128;
  const int tid = threadIdx.x;
  const int lane = tid & 31;
  const int wave = tid >> 5;
  const int wr = wave & 3;    // 32-row chunk
  const int wc = wave >> 2;   // 64-col chunk

  v8f acc[2][4];
#pragma unroll
  for (int ms = 0; ms < 2; ++ms)
#pragma unroll
    for (int nt = 0; nt < 4; ++nt)
#pragma unroll
      for (int r = 0; r < 8; ++r) acc[ms][nt][r] = 0.0f;

  // stage one 128x32 A tile + 128x32 W tile into LDS buffer `buf`
  auto stage = [&](int buf, int k0) {
    if (A_IS_BF16) {
      if (wave == 0) {
        const __bf16* A = (const __bf16*)Aptr;
        tdm_load_2d_bf16(&As[buf][0][0], A + (size_t)m0 * K + k0,
                         /*tile_w=*/32, /*tile_h=*/128,
                         /*row_stride=*/(unsigned)K);
      }
    } else {
      const float* A = (const float*)Aptr;
#pragma unroll
      for (int i = 0; i < 4; ++i) {
        int idx = tid + i * 256;          // 1024 chunks of 4 f32
        int r = idx >> 3;                 // 8 chunks per row
        int c = (idx & 7) * 4;
        const float* src = A + (size_t)(m0 + r) * K + k0 + c;
        const float4 vdat = *(const float4*)src;
        As[buf][r][c + 0] = f2bf(vdat.x);
        As[buf][r][c + 1] = f2bf(vdat.y);
        As[buf][r][c + 2] = f2bf(vdat.z);
        As[buf][r][c + 3] = f2bf(vdat.w);
        if (k0 + 32 < K) __builtin_prefetch(src + 32, 0, 0);
      }
    }
#pragma unroll
    for (int i = 0; i < 4; ++i) {
      int idx = tid + i * 256;
      int r = idx >> 3;
      int c = (idx & 7) * 4;
      const float* src = W + (size_t)(n0 + r) * K + k0 + c;
      const float4 vdat = *(const float4*)src;
      Ws[buf][r][c + 0] = f2bf(vdat.x);
      Ws[buf][r][c + 1] = f2bf(vdat.y);
      Ws[buf][r][c + 2] = f2bf(vdat.z);
      Ws[buf][r][c + 3] = f2bf(vdat.w);
      if (k0 + 32 < K) __builtin_prefetch(src + 32, 0, 0);
    }
  };

  const int nks = K / 32;
  stage(0, 0);
  for (int ks = 0; ks < nks; ++ks) {
    const int cur = ks & 1;
    if (A_IS_BF16 && wave == 0) __builtin_amdgcn_s_wait_tensorcnt(0);
    __syncthreads();                      // tile `cur` ready for everyone
    if (ks + 1 < nks) stage(cur ^ 1, (ks + 1) * 32);  // overlap next tile

    // ---- A fragments: 16x32, ISA layout (lane<16: row=lane, K 0-7 / 16-23;
    //      lane>=16: row=lane-16, K 8-15 / 24-31) ----
    v16bf afrag[2];
#pragma unroll
    for (int ms = 0; ms < 2; ++ms) {
      const int am = wr * 32 + ms * 16 + (lane & 15);
      const int kb = (lane >> 4) * 8;
#pragma unroll
      for (int e = 0; e < 8; ++e) {
        afrag[ms][e]     = As[cur][am][kb + e];
        afrag[ms][8 + e] = As[cur][am][16 + kb + e];
      }
    }
    // ---- B fragments: 32x16; lane<16: col=lane, K 0-7 / 16-23;
    //      lane>=16: col=lane-16, K 8-15 / 24-31.  Each reused by 2 WMMAs. ----
#pragma unroll
    for (int nt = 0; nt < 4; ++nt) {
      v16bf bfrag;
      const int bn = wc * 64 + nt * 16 + (lane & 15);
      const int kb = (lane >> 4) * 8;
#pragma unroll
      for (int e = 0; e < 8; ++e) {
        bfrag[e] = Ws[cur][bn][kb + e];
        bfrag[8 + e] = Ws[cur][bn][16 + kb + e];
      }
#pragma unroll
      for (int ms = 0; ms < 2; ++ms)
        acc[ms][nt] = __builtin_amdgcn_wmma_f32_16x16x32_bf16(
            false, afrag[ms], false, bfrag, (short)0, acc[ms][nt], false,
            false);
    }
  }

  // ---- epilogue: D layout row = r + 8*(lane>=16), col = lane&15 ----
  const int half = lane >> 4;
  const int ncol = lane & 15;
#pragma unroll
  for (int nt = 0; nt < 4; ++nt) {
    const int n = n0 + wc * 64 + nt * 16 + ncol;
    const float bv = bias[n];
#pragma unroll
    for (int ms = 0; ms < 2; ++ms)
#pragma unroll
      for (int r = 0; r < 8; ++r) {
        const int m = m0 + wr * 32 + ms * 16 + r + 8 * half;
        const float val = acc[ms][nt][r] + bv;
        if (OUT_BF16)
          ((__bf16*)Out)[(size_t)m * N + n] = f2bf(val);
        else
          ((float*)Out)[(size_t)m * N + n] = val;
      }
  }
}

// ---------------------------------------------------------------------------
// RoPE (in place on bf16 q or k): x layout (B, S, heads, HD)
// freqs_cis layout (S, HD/2, 2) = (cos, sin)
// ---------------------------------------------------------------------------
__global__ __launch_bounds__(256) void rope_kernel(
    __bf16* __restrict__ x, const float* __restrict__ fc,
    const int* __restrict__ startpos, int heads) {
  const int HD2 = HEADDIM / 2;
  const size_t total = (size_t)BATCH * SEQ * heads * HD2;
  size_t i = (size_t)blockIdx.x * blockDim.x + threadIdx.x;
  if (i >= total) return;
  const int d2 = (int)(i % HD2);
  const int s = (int)((i / ((size_t)HD2 * heads)) % SEQ);
  const float2 cs = ((const float2*)fc)[(size_t)(startpos[0] + s) * HD2 + d2];
  const float x0 = (float)x[2 * i];
  const float x1 = (float)x[2 * i + 1];
  x[2 * i]     = f2bf(x0 * cs.x - x1 * cs.y);
  x[2 * i + 1] = f2bf(x0 * cs.y + x1 * cs.x);
}

// ---------------------------------------------------------------------------
// Flash attention (causal, GQA).
// grid = (S/64, QHEADS, BATCH), block = 128 (4 waves).
// Each wave owns 16 query rows; block owns 64. KV processed in 32-key tiles.
// K tile is copied global->LDS with CDNA5 async-to-LDS ops; V is transposed
// into LDS through VGPRs (needed for the B-operand layout of PV).
// q: (B,S,QHEADS,HD) bf16 ; k,v: (B,S,KVHEADS,HD) bf16 ; ctx out same as q.
// ---------------------------------------------------------------------------
__global__ __launch_bounds__(128) void flash_attn_kernel(
    const __bf16* __restrict__ q, const __bf16* __restrict__ k,
    const __bf16* __restrict__ v, __bf16* __restrict__ ctx) {
  __shared__ __bf16 Ks[32][HEADDIM];       // keys x d
  __shared__ __bf16 Vt[HEADDIM][32];       // d x keys (transposed for B-frag)
  __shared__ __bf16 Ps[4][16][32];         // per-wave P tile (rows x keys)

  const int qt = blockIdx.x;
  const int h = blockIdx.y;
  const int b = blockIdx.z;
  const int kvh = h / GROUP;
  const int tid = threadIdx.x;
  const int lane = tid & 31;
  const int wave = tid >> 5;
  const int half = lane >> 4;
  const int ncol = lane & 15;

  const float scale = 0.08838834764831845f;  // 1/sqrt(128)
  const int q0 = qt * 64;
  const int qrow_wave = q0 + wave * 16;      // first row this wave owns

  // ---- load Q fragments once: 16 rows x 128 d = 4 A-frags over d ----
  v16bf qa[4];
  {
    const int m = qrow_wave + (lane & 15);
    const __bf16* qrow = q + ((size_t)(b * SEQ + m)) * DMODEL + h * HEADDIM;
    const int kb = (lane >> 4) * 8;
#pragma unroll
    for (int c = 0; c < 4; ++c)
#pragma unroll
      for (int e = 0; e < 8; ++e) {
        qa[c][e]     = qrow[c * 32 + kb + e];
        qa[c][8 + e] = qrow[c * 32 + 16 + kb + e];
      }
  }

  v8f o[8];
  float mrow[8], lrow[8];
#pragma unroll
  for (int j = 0; j < 8; ++j)
#pragma unroll
    for (int r = 0; r < 8; ++r) o[j][r] = 0.0f;
#pragma unroll
  for (int r = 0; r < 8; ++r) { mrow[r] = -INFINITY; lrow[r] = 0.0f; }

  const int nkb = qt * 2 + 2;  // enough kv blocks to cover rows < q0+64
  for (int kb = 0; kb < nkb; ++kb) {
    const int t0 = kb * 32;
    // ---- cooperative load: K tile (32x128) async, V tile transposed ----
#pragma unroll
    for (int i = 0; i < 4; ++i) {
      int idx = tid + i * 128;          // 512 chunks of 8
      int r = idx >> 4;                 // 16 chunks per row
      int c = (idx & 15) * 8;
      const size_t rowoff =
          ((size_t)(b * SEQ + t0 + r)) * KVDIM + kvh * HEADDIM + c;
      async_copy_b128(&Ks[r][c], k + rowoff);
      const __bf16* vs = v + rowoff;
#pragma unroll
      for (int e = 0; e < 8; ++e) Vt[c + e][r] = vs[e];
      if (kb + 1 < nkb) {
        __builtin_prefetch(k + rowoff + (size_t)32 * KVDIM, 0, 0);
        __builtin_prefetch(v + rowoff + (size_t)32 * KVDIM, 0, 0);
      }
    }
    wait_asynccnt0();
    __syncthreads();

    // ---- logits 16x32 = 2 WMMA accumulators, contraction over d=128 ----
    v8f lg[2];
#pragma unroll
    for (int nt = 0; nt < 2; ++nt)
#pragma unroll
      for (int r = 0; r < 8; ++r) lg[nt][r] = 0.0f;
#pragma unroll
    for (int c = 0; c < 4; ++c) {
#pragma unroll
      for (int nt = 0; nt < 2; ++nt) {
        v16bf bfrag;
        const int key = nt * 16 + (lane & 15);
        const int dkb = (lane >> 4) * 8;
#pragma unroll
        for (int e = 0; e < 8; ++e) {
          bfrag[e] = Ks[key][c * 32 + dkb + e];
          bfrag[8 + e] = Ks[key][c * 32 + 16 + dkb + e];
        }
        lg[nt] = __builtin_amdgcn_wmma_f32_16x16x32_bf16(
            false, qa[c], false, bfrag, (short)0, lg[nt], false, false);
      }
    }

    // ---- causal mask + online softmax (row = r + 8*half, col = ncol) ----
    float alpha[8];
#pragma unroll
    for (int r = 0; r < 8; ++r) {
      const int mrow_g = qrow_wave + r + 8 * half;   // global query row
      float x0 = lg[0][r] * scale;
      float x1 = lg[1][r] * scale;
      if (t0 + ncol > mrow_g) x0 = -1e30f;
      if (t0 + 16 + ncol > mrow_g) x1 = -1e30f;
      float mx = fmaxf(x0, x1);
#pragma unroll
      for (int s = 1; s < 16; s <<= 1) mx = fmaxf(mx, __shfl_xor(mx, s, 32));
      const float mnew = fmaxf(mrow[r], mx);
      const float p0 = __expf(x0 - mnew);
      const float p1 = __expf(x1 - mnew);
      float sm = p0 + p1;
#pragma unroll
      for (int s = 1; s < 16; s <<= 1) sm += __shfl_xor(sm, s, 32);
      alpha[r] = __expf(mrow[r] - mnew);
      lrow[r] = lrow[r] * alpha[r] + sm;
      mrow[r] = mnew;
      const int mloc = r + 8 * half;
      Ps[wave][mloc][ncol] = f2bf(p0);
      Ps[wave][mloc][16 + ncol] = f2bf(p1);
    }
    // rescale running output
#pragma unroll
    for (int j = 0; j < 8; ++j)
#pragma unroll
      for (int r = 0; r < 8; ++r) o[j][r] *= alpha[r];
    __syncthreads();  // P visible (and Ks/Vt reads done)

    // ---- PV: A = P (16x32), B = V chunk (32 keys x 16 d), 8 d-chunks ----
    v16bf pa;
    {
      const int m = lane & 15;
      const int kbp = (lane >> 4) * 8;
#pragma unroll
      for (int e = 0; e < 8; ++e) {
        pa[e]     = Ps[wave][m][kbp + e];
        pa[8 + e] = Ps[wave][m][16 + kbp + e];
      }
    }
#pragma unroll
    for (int j = 0; j < 8; ++j) {
      v16bf vb;
      const int n = lane & 15;
      const int tb = (lane >> 4) * 8;
#pragma unroll
      for (int e = 0; e < 8; ++e) {
        vb[e] = Vt[j * 16 + n][tb + e];
        vb[8 + e] = Vt[j * 16 + n][16 + tb + e];
      }
      o[j] = __builtin_amdgcn_wmma_f32_16x16x32_bf16(
          false, pa, false, vb, (short)0, o[j], false, false);
    }
    __syncthreads();  // done with Ks/Vt/Ps before next tile overwrites
  }

  // ---- epilogue: normalize and store ctx (B,S,QHEADS,HD) ----
#pragma unroll
  for (int j = 0; j < 8; ++j)
#pragma unroll
    for (int r = 0; r < 8; ++r) {
      const int m = qrow_wave + r + 8 * half;
      const float val = o[j][r] / lrow[r];
      ctx[((size_t)(b * SEQ + m)) * DMODEL + h * HEADDIM + j * 16 + ncol] =
          f2bf(val);
    }
}

// ---------------------------------------------------------------------------
// Launch: inputs (in order): xs, startpos, freqs_cis, Wq, bq, Wk, bk,
//                            Wv, bv, Wo, bo
// Workspace layout (bf16): q (B*S*D) | k (B*S*KVD) | v (B*S*KVD) | ctx (B*S*D)
// = ~84 MB.
// ---------------------------------------------------------------------------
extern "C" void kernel_launch(void* const* d_in, const int* in_sizes, int n_in,
                              void* d_out, int out_size, void* d_ws,
                              size_t ws_size, hipStream_t stream) {
  const float* xs = (const float*)d_in[0];
  const int* startpos = (const int*)d_in[1];
  const float* fc = (const float*)d_in[2];
  const float* Wq = (const float*)d_in[3];
  const float* bq = (const float*)d_in[4];
  const float* Wk = (const float*)d_in[5];
  const float* bk = (const float*)d_in[6];
  const float* Wv = (const float*)d_in[7];
  const float* bv = (const float*)d_in[8];
  const float* Wo = (const float*)d_in[9];
  const float* bo = (const float*)d_in[10];
  float* out = (float*)d_out;

  const size_t M = (size_t)BATCH * SEQ;  // 4096 token rows
  __bf16* qb = (__bf16*)d_ws;
  __bf16* kb = qb + M * DMODEL;
  __bf16* vb = kb + M * KVDIM;
  __bf16* cb = vb + M * KVDIM;

  const dim3 blk256(256);
  const dim3 blk128(128);

  // QKV projections (f32 in -> bf16 out, converted while staging LDS)
  gemm_wt_kernel<false, true>
      <<<dim3(M / 128, DMODEL / 128), blk256, 0, stream>>>(
          xs, Wq, bq, qb, (int)M, DMODEL, DMODEL);
  gemm_wt_kernel<false, true>
      <<<dim3(M / 128, KVDIM / 128), blk256, 0, stream>>>(
          xs, Wk, bk, kb, (int)M, KVDIM, DMODEL);
  gemm_wt_kernel<false, true>
      <<<dim3(M / 128, KVDIM / 128), blk256, 0, stream>>>(
          xs, Wv, bv, vb, (int)M, KVDIM, DMODEL);

  // RoPE on q and k
  {
    const size_t qpairs = M * QHEADS * (HEADDIM / 2);
    const size_t kpairs = M * KVHEADS * (HEADDIM / 2);
    rope_kernel<<<dim3((qpairs + 255) / 256), dim3(256), 0, stream>>>(
        qb, fc, startpos, QHEADS);
    rope_kernel<<<dim3((kpairs + 255) / 256), dim3(256), 0, stream>>>(
        kb, fc, startpos, KVHEADS);
  }

  // Flash attention
  flash_attn_kernel<<<dim3(SEQ / 64, QHEADS, BATCH), blk128, 0, stream>>>(
      qb, kb, vb, cb);

  // Output projection (bf16 ctx -> f32 out, A tile staged via TDM)
  gemm_wt_kernel<true, false>
      <<<dim3(M / 128, DMODEL / 128), blk256, 0, stream>>>(
          cb, Wo, bo, out, (int)M, DMODEL, DMODEL);
}
